// ConvMHSA_18150531793497
// MI455X (gfx1250) — matrix operand order repaired and checked by reading, not compiled, not run on hardware
//
#include <hip/hip_runtime.h>
#include <hip/hip_bf16.h>

typedef __attribute__((ext_vector_type(16))) __bf16 v16bf;
typedef __attribute__((ext_vector_type(8)))  __bf16 v8bf;
typedef __attribute__((ext_vector_type(8)))  float  v8f;
typedef int v4i_vs __attribute__((vector_size(4 * sizeof(int))));  // b128 unit

#define NB    64      // batch
#define CIN   64
#define HH    64
#define WW    128
#define NPIX  8192            // H*W
#define KCONV 576             // CIN*9, reordered as tap*64+cin
#define KATT  131072          // D_K * NPIX
#define NHEAD 8

#define AS1 __attribute__((address_space(1)))
#define AS3 __attribute__((address_space(3)))

#if defined(__has_builtin)
#if __has_builtin(__builtin_amdgcn_global_load_async_to_lds_b128)
#define HAVE_ASYNC_LDS 1
#endif
#endif

// ---------------------------------------------------------------- WMMA helpers

__device__ inline v8f wmma_bf16(v16bf a, v16bf b, v8f c) {
  return __builtin_amdgcn_wmma_f32_16x16x32_bf16(
      false, a, false, b, (short)0, c, false, false);
}

// A-matrix (16x32 bf16, MxK): lane L holds row M=L&15.
// elems 0..7 -> K=k0+off+e ; elems 8..15 -> K=k0+16+off+e (off = L<16 ? 0 : 8)
__device__ inline v16bf load_a_frag(const __bf16* base, int m_row, long ldk,
                                    int k0, int lane) {
  const __bf16* p = base + (size_t)m_row * (size_t)ldk + k0 + ((lane & 16) ? 8 : 0);
  v8bf lo = *(const v8bf*)(p);
  v8bf hi = *(const v8bf*)(p + 16);
  v16bf f;
#pragma unroll
  for (int e = 0; e < 8; ++e) { f[e] = lo[e]; f[8 + e] = hi[e]; }
  return f;
}

// Pack A fragment from two preloaded 16B chunks (LDS case)
__device__ inline v16bf pack_frag(v8bf lo, v8bf hi) {
  v16bf f;
#pragma unroll
  for (int e = 0; e < 8; ++e) { f[e] = lo[e]; f[8 + e] = hi[e]; }
  return f;
}

// B-matrix (32x16 bf16, KxN): lane L holds column N=L&15,
// elems 0..7 -> K=k0+off+e ; elems 8..15 -> K=k0+16+off+e (off = L<16 ? 0 : 8)
__device__ inline v16bf load_b_frag(const __bf16* base, int n_row, long ldk,
                                    int k0, int lane) {
  const __bf16* p = base + (size_t)n_row * (size_t)ldk + k0 + ((lane & 16) ? 8 : 0);
  v8bf lo = *(const v8bf*)(p);
  v8bf hi = *(const v8bf*)(p + 16);
  return pack_frag(lo, hi);
}

// ---------------------------------------------------------------- weight prep
// Conv weights: f32 [cout][cin][3][3] -> bf16 [cout][tap*64+cin] (tap-major K).
// W_out (128x64 f32) -> bf16 transposed [64][128].
__global__ __launch_bounds__(256) void prep_weights(
    const float* __restrict__ Wqk, const float* __restrict__ Wv,
    const float* __restrict__ Wout,
    __bf16* __restrict__ Wqk_b, __bf16* __restrict__ Wv_b,
    __bf16* __restrict__ Wout_t) {
  int idx = blockIdx.x * 256 + threadIdx.x;
  if (idx < 256 * KCONV) {
    int cout = idx / KCONV, r = idx % KCONV;
    int tap = r >> 6, cin = r & 63;
    Wqk_b[idx] = (__bf16)Wqk[cout * KCONV + cin * 9 + tap];
  }
  if (idx < 128 * KCONV) {
    int cout = idx / KCONV, r = idx % KCONV;
    int tap = r >> 6, cin = r & 63;
    Wv_b[idx] = (__bf16)Wv[cout * KCONV + cin * 9 + tap];
  }
  if (idx < 128 * 64) {
    int kc = idx >> 6, dout = idx & 63;
    Wout_t[dout * 128 + kc] = (__bf16)Wout[idx];
  }
}

// ---------------------------------------------------------------- x prep
// x f32 [n][c][h][w] -> bf16 channel-minor x_t[n][h][w][c] via LDS transpose.
__global__ __launch_bounds__(256) void prep_x(const float* __restrict__ x,
                                              __bf16* __restrict__ xt) {
  const int gi = blockIdx.x, n = blockIdx.y;
  __shared__ __align__(16) __bf16 t[WW][72];  // pad 64->72 (16B-aligned rows)
  for (int idx = threadIdx.x; idx < CIN * WW; idx += 256) {
    int c = idx >> 7, gj = idx & 127;  // consecutive gj -> coalesced reads
    t[gj][c] = (__bf16)x[(((size_t)n * CIN + c) * HH + gi) * WW + gj];
  }
  __syncthreads();
  for (int idx = threadIdx.x; idx < WW * 8; idx += 256) {
    int gj = idx >> 3, c0 = (idx & 7) * 8;
    *(v8bf*)(xt + (((size_t)n * HH + gi) * WW + gj) * CIN + c0) =
        *(const v8bf*)&t[gj][c0];
  }
}

// ---------------------------------------------------------------- conv (implicit GEMM)
// One workgroup: image n, row i, 64 output channels (block cb), 128 columns.
// M = 128 pixels (8 waves x 16), N = 64 channels (4 tiles), K = 576 tap-major.
// mode 0: qk conv -> q[h][n][P*16+c], k[h][m][P*16+c]
// mode 1: v conv  -> vt[h][P*16+d][m]  (m-contiguous for the softA*V GEMM)
__global__ __launch_bounds__(256) void conv_qkv_wmma(
    const __bf16* __restrict__ xt, const __bf16* __restrict__ Wb,
    const float* __restrict__ bias,
    __bf16* __restrict__ q, __bf16* __restrict__ kbuf, __bf16* __restrict__ vt,
    int mode) {
  const int cb = blockIdx.x;
  const int i  = blockIdx.y;
  const int n  = blockIdx.z;
  const int tid  = threadIdx.x;
  const int lane = tid & 31;
  const int wave = tid >> 5;

  // channel-minor patch cache: [halo row][col jj (gj=jj-1)][cin], padded to 72
  __shared__ __align__(16) __bf16 xs[3][132][72];

  // ---- stage 3x130x64 bf16 tile; pure 16B copies (async-to-LDS if available)
  bool pending_async = false;
  for (int t = tid; t < 3 * 130 * 8; t += 256) {
    int r   = t / (130 * 8);
    int rem = t % (130 * 8);
    int jj  = rem >> 3;
    int c0  = (rem & 7) * 8;
    int gi = i + r - 1, gj = jj - 1;
    __bf16* ldst = &xs[r][jj][c0];
    if (gi >= 0 && gi < HH && gj >= 0 && gj < WW) {
      const __bf16* src = xt + (((size_t)n * HH + gi) * WW + gj) * CIN + c0;
#if HAVE_ASYNC_LDS
      __builtin_amdgcn_global_load_async_to_lds_b128(
          (AS1 v4i_vs*)src, (AS3 v4i_vs*)ldst, 0, 0);
      pending_async = true;
#else
      *(v8bf*)ldst = *(const v8bf*)src;
#endif
    } else {
      v8bf z = {};
      *(v8bf*)ldst = z;
    }
  }
#if HAVE_ASYNC_LDS
  if (pending_async) {
#if __has_builtin(__builtin_amdgcn_s_wait_asynccnt)
    __builtin_amdgcn_s_wait_asynccnt(0);
#else
    asm volatile("s_wait_asynccnt 0" ::: "memory");
#endif
  }
#else
  (void)pending_async;
#endif
  __syncthreads();

  const int pj  = wave * 16 + (lane & 15);  // pixel column (A row)
  const int off = (lane & 16) ? 8 : 0;

  v8f acc[4] = {};
#pragma unroll
  for (int tap = 0; tap < 9; ++tap) {
    const int ky = tap / 3, kx = tap % 3;
    const __bf16* xrow = &xs[ky][pj + kx][0];
#pragma unroll
    for (int half = 0; half < 2; ++half) {
      const int cin0 = half * 32;
      v8bf lo = *(const v8bf*)(xrow + cin0 + off);
      v8bf hi = *(const v8bf*)(xrow + cin0 + off + 16);
      v16bf a = pack_frag(lo, hi);
      const int kc0 = tap * 64 + cin0;
#pragma unroll
      for (int nt = 0; nt < 4; ++nt) {
        int cout_row = cb * 64 + nt * 16 + (lane & 15);
        v16bf b = load_b_frag(Wb, cout_row, KCONV, kc0, lane);
        acc[nt] = wmma_bf16(a, b, acc[nt]);
      }
    }
  }

  const int mhi = (lane & 16) ? 8 : 0;
#pragma unroll
  for (int nt = 0; nt < 4; ++nt) {
    int cout = cb * 64 + nt * 16 + (lane & 15);
    float bs = bias[cout];
#pragma unroll
    for (int j = 0; j < 8; ++j) {
      int pix = wave * 16 + j + mhi;
      size_t P = (size_t)i * WW + pix;
      __bf16 bv = (__bf16)(acc[nt][j] + bs);
      if (mode == 0) {
        int h = cout >> 5, s = (cout >> 4) & 1, c = cout & 15;
        __bf16* dst = s ? kbuf : q;
        dst[((size_t)h * NB + n) * KATT + P * 16 + c] = bv;
      } else {
        int h = cout >> 4, d = cout & 15;
        vt[((size_t)h * KATT + P * 16 + d) * NB + n] = bv;
      }
    }
  }
}

// ---------------------------------------------------------------- attention scores (partial)
// Per (split s, head h): partial A[n,m] over a 2048-wide K-slice; each wave owns
// a 256-wide sub-slice and the full 64x64 tile grid -> Apart[s*8+wave][h][n][m].
__global__ __launch_bounds__(256) void scores_partial_wmma(
    const __bf16* __restrict__ q, const __bf16* __restrict__ kk,
    float* __restrict__ Apart) {
  const int s = blockIdx.x;   // 0..63
  const int h = blockIdx.y;   // 0..7
  const int lane = threadIdx.x & 31;
  const int wave = threadIdx.x >> 5;

  const __bf16* qh = q  + (size_t)h * NB * KATT;
  const __bf16* kh = kk + (size_t)h * NB * KATT;
  const int kbase = s * 2048 + wave * 256;

  v8f acc[4][4] = {};
  for (int kc = 0; kc < 256; kc += 32) {
    int k0 = kbase + kc;
    v16bf a[4], b[4];
#pragma unroll
    for (int mt = 0; mt < 4; ++mt)
      a[mt] = load_a_frag(qh, mt * 16 + (lane & 15), KATT, k0, lane);
#pragma unroll
    for (int nt = 0; nt < 4; ++nt)
      b[nt] = load_b_frag(kh, nt * 16 + (lane & 15), KATT, k0, lane);
#pragma unroll
    for (int mt = 0; mt < 4; ++mt)
#pragma unroll
      for (int nt = 0; nt < 4; ++nt)
        acc[mt][nt] = wmma_bf16(a[mt], b[nt], acc[mt][nt]);
  }

  float* dst = Apart + ((size_t)(s * 8 + wave) * NHEAD + h) * 4096;
  const int mhi = (lane & 16) ? 8 : 0;
#pragma unroll
  for (int mt = 0; mt < 4; ++mt)
#pragma unroll
    for (int nt = 0; nt < 4; ++nt)
#pragma unroll
      for (int j = 0; j < 8; ++j)
        dst[(mt * 16 + j + mhi) * 64 + nt * 16 + (lane & 15)] = acc[mt][nt][j];
}

// ---------------------------------------------------------------- reduce + softmax
// One block per (h,n); 64 threads = one m each. Sequential f32 reduction over
// 512 partial slices -> bitwise-deterministic across graph replays.
__global__ __launch_bounds__(64) void softmax_rows(
    const float* __restrict__ Apart, __bf16* __restrict__ Pm) {
  const int n = blockIdx.x, h = blockIdx.y;
  const int m = threadIdx.x;
  float a = 0.0f;
  for (int s = 0; s < 512; ++s)
    a += Apart[(((size_t)s * NHEAD + h) * 64 + n) * 64 + m];
  a *= (1.0f / 362.03867196751236f);  // 1/sqrt(D_K*H*W)

  __shared__ float buf[64], buf2[64];
  buf[m] = a;
  __syncthreads();
  float mx = -3.0e38f;
  for (int t = 0; t < 64; ++t) mx = fmaxf(mx, buf[t]);
  float e = __expf(a - mx);
  buf2[m] = e;
  __syncthreads();
  float sum = 0.0f;
  for (int t = 0; t < 64; ++t) sum += buf2[t];
  Pm[((size_t)h * 64 + n) * 64 + m] = (__bf16)(e / sum);
}

// ---------------------------------------------------------------- softA x V
// Per (head h, pc-tile of 1024): sa[n, pc] = sum_m P[h][n][m] * vt[h][pc][m].
// Output layout sa[n][P][128ch] (ch = h*16+d): contiguous K for the projection.
__global__ __launch_bounds__(256) void attn_apply_wmma(
    const __bf16* __restrict__ Pm, const __bf16* __restrict__ vt,
    __bf16* __restrict__ sa) {
  const int tile = blockIdx.x;  // 0..127
  const int h    = blockIdx.y;
  const int lane = threadIdx.x & 31;
  const int wave = threadIdx.x >> 5;
  const int mt = wave & 3;
  const int ng = wave >> 2;

  const __bf16* Ph = Pm + (size_t)h * 64 * 64;
  const __bf16* vh = vt + (size_t)h * KATT * NB;

  v16bf a0 = load_a_frag(Ph, mt * 16 + (lane & 15), 64, 0, lane);
  v16bf a1 = load_a_frag(Ph, mt * 16 + (lane & 15), 64, 32, lane);

  const int mhi = (lane & 16) ? 8 : 0;
  for (int nt = 0; nt < 32; ++nt) {
    int pc = tile * 1024 + (ng * 32 + nt) * 16 + (lane & 15);
    v16bf b0 = load_b_frag(vh, pc, NB, 0, lane);
    v16bf b1 = load_b_frag(vh, pc, NB, 32, lane);
    v8f acc = {};
    acc = wmma_bf16(a0, b0, acc);
    acc = wmma_bf16(a1, b1, acc);
    int Ppix = pc >> 4, d = pc & 15;
#pragma unroll
    for (int j = 0; j < 8; ++j) {
      int nrow = mt * 16 + j + mhi;
      sa[((size_t)nrow * NPIX + Ppix) * 128 + h * 16 + d] = (__bf16)acc[j];
    }
  }
}

// ---------------------------------------------------------------- output projection
// out[n][dout][P] = sum_kc sa[n][P][kc] * W_out[kc][dout] + b_out[dout]
__global__ __launch_bounds__(256) void out_proj_wmma(
    const __bf16* __restrict__ sa, const __bf16* __restrict__ Wot,
    const float* __restrict__ bout, float* __restrict__ out) {
  const int tile = blockIdx.x;  // 0..4095
  const int lane = threadIdx.x & 31;
  const int wave = threadIdx.x >> 5;
  const int rbase = tile * 128 + wave * 16;

  v16bf a[4];
#pragma unroll
  for (int kc = 0; kc < 4; ++kc)
    a[kc] = load_a_frag(sa, rbase + (lane & 15), 128, kc * 32, lane);

  const int mhi = (lane & 16) ? 8 : 0;
#pragma unroll
  for (int nt = 0; nt < 4; ++nt) {
    int dout = nt * 16 + (lane & 15);
    v8f acc = {};
#pragma unroll
    for (int kc = 0; kc < 4; ++kc) {
      v16bf b = load_b_frag(Wot, dout, 128, kc * 32, lane);
      acc = wmma_bf16(a[kc], b, acc);
    }
    float bs = bout[dout];
#pragma unroll
    for (int j = 0; j < 8; ++j) {
      int r = rbase + j + mhi;
      int n = r >> 13;
      int Ppix = r & (NPIX - 1);
      out[((size_t)n * 64 + dout) * NPIX + Ppix] = acc[j] + bs;
    }
  }
}

// ---------------------------------------------------------------- launcher
extern "C" void kernel_launch(void* const* d_in, const int* in_sizes, int n_in,
                              void* d_out, int out_size, void* d_ws, size_t ws_size,
                              hipStream_t stream) {
  (void)in_sizes; (void)n_in; (void)out_size; (void)ws_size;
  const float* x     = (const float*)d_in[0];
  const float* W_qk  = (const float*)d_in[1];
  const float* b_qk  = (const float*)d_in[2];
  const float* W_v   = (const float*)d_in[3];
  const float* b_v   = (const float*)d_in[4];
  const float* W_out = (const float*)d_in[5];
  const float* b_out = (const float*)d_in[6];
  float* out = (float*)d_out;

  char* ws = (char*)d_ws;
  const size_t SZ_QKV = (size_t)NHEAD * NB * KATT * sizeof(__bf16);  // 134 MB each
  __bf16* q    = (__bf16*)(ws);
  __bf16* kbuf = (__bf16*)(ws + SZ_QKV);
  __bf16* vt   = (__bf16*)(ws + 2 * SZ_QKV);
  __bf16* Pm   = (__bf16*)(ws + 3 * SZ_QKV);                 // 64 KB
  // x_t (bf16, 64 MB) is dead once the convs finish; Apart (f32, 64 MB) is
  // first written by the score kernel afterwards -> alias the same region.
  char*   shared_region = ws + 3 * SZ_QKV + 65536;
  __bf16* xt    = (__bf16*)shared_region;
  float*  Apart = (float*)shared_region;
  char*   wbase = shared_region + (size_t)512 * NHEAD * 4096 * 4;
  __bf16* Wqk_b  = (__bf16*)(wbase);
  __bf16* Wv_b   = (__bf16*)(wbase + 256 * KCONV * 2);
  __bf16* Wout_t = (__bf16*)(wbase + (256 + 128) * KCONV * 2);
  __bf16* sa = q;  // q is dead after the score GEMM; reuse for sa

  prep_weights<<<576, 256, 0, stream>>>(W_qk, W_v, W_out, Wqk_b, Wv_b, Wout_t);
  dim3 gx(HH, NB);
  prep_x<<<gx, 256, 0, stream>>>(x, xt);

  dim3 gqk(4, HH, NB);
  conv_qkv_wmma<<<gqk, 256, 0, stream>>>(xt, Wqk_b, b_qk, q, kbuf, vt, 0);
  dim3 gv(2, HH, NB);
  conv_qkv_wmma<<<gv, 256, 0, stream>>>(xt, Wv_b, b_v, q, kbuf, vt, 1);

  dim3 gsc(64, NHEAD);
  scores_partial_wmma<<<gsc, 256, 0, stream>>>(q, kbuf, Apart);

  dim3 gsm(64, NHEAD);
  softmax_rows<<<gsm, 64, 0, stream>>>(Apart, Pm);

  dim3 gsa(128, NHEAD);
  attn_apply_wmma<<<gsa, 256, 0, stream>>>(Pm, vt, sa);

  out_proj_wmma<<<4096, 256, 0, stream>>>(sa, Wout_t, b_out, out);
}
